// HAN_87265145520188
// MI455X (gfx1250) — matrix operand order repaired and checked by reading, not compiled, not run on hardware
//
#include <hip/hip_runtime.h>
#include <hip/hip_bf16.h>
#include <math.h>

#define HEADS 8
#define HID   256
#define DH    32
#define NA    10000
#define NPAP  20000
#define CIN   256
#define NE    256000
#define OUTF  16

typedef __bf16 v16bf __attribute__((ext_vector_type(16)));
typedef float  v8f   __attribute__((ext_vector_type(8)));

// ---------------------------------------------------------------------------
// Monotonic float<->uint encoding for atomicMax-based segment max
// ---------------------------------------------------------------------------
__device__ __forceinline__ unsigned fenc(float f) {
    unsigned u = __float_as_uint(f);
    return (u & 0x80000000u) ? ~u : (u | 0x80000000u);
}
__device__ __forceinline__ float fdec(unsigned u) {
    unsigned b = (u & 0x80000000u) ? (u & 0x7FFFFFFFu) : ~u;
    return __uint_as_float(b);
}

// ---------------------------------------------------------------------------
// Async global->LDS copy of one 32x256 f32 k-panel of W (32 KB).
// 256 threads x 8 float4 each; tracked on ASYNCcnt (8 wave-instrs per wave).
// ldsBase = LDS byte offset of the destination buffer (low 32 bits of the
// flat pointer: the LDS aperture maps addr[31:0] directly to the LDS offset).
// ---------------------------------------------------------------------------
__device__ __forceinline__ void async_copy_panel(const float* __restrict__ W,
                                                 int k0, unsigned ldsBase, int t)
{
    #pragma unroll
    for (int i = 0; i < 8; ++i) {
        const int idx = t + 256 * i;          // float4 index within the panel
        const int kk  = idx >> 6;             // panel row (0..31)
        const int col = (idx & 63) * 4;       // column (0..252)
        const unsigned lds = ldsBase + (unsigned)idx * 16u;
        const unsigned long long ga =
            (unsigned long long)(const void*)(W + (size_t)(k0 + kk) * HID + col);
        asm volatile("global_load_async_to_lds_b128 %0, %1, off"
                     :: "v"(lds), "v"(ga) : "memory");
    }
}

// ---------------------------------------------------------------------------
// WMMA GEMM: C[M,256] = A[M,256] @ W[256,256] + bias   (bf16 in, f32 acc)
// One wave -> one 16x32 strip of C (two 16x16 WMMA tiles); 8 waves/block
// cover all 256 output columns; grid.x covers M/16 row tiles (M % 16 == 0).
// W k-panels are staged in LDS via async copies, double buffered so the copy
// of panel k+1 overlaps the WMMA work on panel k.
// ---------------------------------------------------------------------------
__global__ __launch_bounds__(256)
void gemm_bias_wmma(const float* __restrict__ A, const float* __restrict__ W,
                    const float* __restrict__ bias, float* __restrict__ C, int M)
{
    __shared__ float sW[2][32 * HID];     // 2 x 32KB double buffer

    const int t    = threadIdx.x;
    const int lane = t & 31;
    const int wave = t >> 5;
    const int r    = lane & 15;
    const int half = lane >> 4;
    const int m0   = blockIdx.x * 16;
    const int row  = m0 + r;              // A-matrix: both lane halves hold M=0..15
    const int n0   = wave * 32;

    const unsigned ldsBuf0 = (unsigned)(unsigned long long)(void*)&sW[0][0];
    const unsigned ldsBuf1 = (unsigned)(unsigned long long)(void*)&sW[1][0];

    v8f acc0 = {};
    v8f acc1 = {};

    async_copy_panel(W, 0, ldsBuf0, t);   // prologue: panel 0 -> buf 0

    for (int kb = 0; kb < CIN / 32; ++kb) {
        const int k0 = kb * 32;
        if (kb + 1 < CIN / 32) {
            async_copy_panel(W, k0 + 32, (kb & 1) ? ldsBuf0 : ldsBuf1, t);
            asm volatile("s_wait_asynccnt 0x8" ::: "memory");   // panel kb done
        } else {
            asm volatile("s_wait_asynccnt 0x0" ::: "memory");
        }
        __syncthreads();                  // panel kb visible to all waves

        if (k0 + 32 < CIN)
            __builtin_prefetch(&A[row * CIN + k0 + 32], 0, 0);

        // A fragment: 16x32, lanes 0-15 K=0..15, lanes 16-31 K=16..31
        v16bf a = {};
        #pragma unroll
        for (int p = 0; p < 8; ++p) {
            const int kp = 2 * p + 16 * half;
            a[2 * p]     = (__bf16)A[row * CIN + k0 + kp];
            a[2 * p + 1] = (__bf16)A[row * CIN + k0 + kp + 1];
        }
        // B fragments from LDS: 32x16, lanes 0-15 K=0..15, lanes 16-31 K=16..31
        const float* __restrict__ buf = sW[kb & 1];
        v16bf b0 = {}, b1 = {};
        #pragma unroll
        for (int j = 0; j < 8; ++j) {
            const int kk = 2 * j + 16 * half;
            b0[2 * j]     = (__bf16)buf[kk * HID + n0 + r];
            b0[2 * j + 1] = (__bf16)buf[(kk + 1) * HID + n0 + r];
            b1[2 * j]     = (__bf16)buf[kk * HID + n0 + 16 + r];
            b1[2 * j + 1] = (__bf16)buf[(kk + 1) * HID + n0 + 16 + r];
        }
        acc0 = __builtin_amdgcn_wmma_f32_16x16x32_bf16(false, a, false, b0,
                                                       (short)0, acc0, false, false);
        acc1 = __builtin_amdgcn_wmma_f32_16x16x32_bf16(false, a, false, b1,
                                                       (short)0, acc1, false, false);

        __syncthreads();                  // done reading buf before it is refilled
    }

    // C layout: VGPR v, lanes 0-15 -> M=v, lanes 16-31 -> M=v+8; N=lane&15
    #pragma unroll
    for (int v = 0; v < 8; ++v) {
        const int rr = m0 + v + 8 * half;
        const int c0 = n0 + r;
        if (rr < M) {
            C[rr * HID + c0]      = acc0[v] + bias[c0];
            C[rr * HID + c0 + 16] = acc1[v] + bias[c0 + 16];
        }
    }
}

// ---------------------------------------------------------------------------
// WMMA GEMM fused with tanh()*q reduction for semantic attention score:
//   score += invN * sum_{n,f} tanh((O @ kW)[n,f] + kb[f]) * q[f]
// ---------------------------------------------------------------------------
__global__ __launch_bounds__(256)
void semantic_score_wmma(const float* __restrict__ O, const float* __restrict__ kW,
                         const float* __restrict__ kb, const float* __restrict__ q,
                         float* __restrict__ score, float invN, int M)
{
    const int lane = threadIdx.x & 31;
    const int wave = threadIdx.x >> 5;
    const int r    = lane & 15;
    const int half = lane >> 4;
    const int m0   = blockIdx.x * 16;
    if (m0 >= M) return;
    const int row  = m0 + r;
    const int n0   = wave * 32;

    v8f acc0 = {};
    v8f acc1 = {};

    for (int k0 = 0; k0 < HID; k0 += 32) {
        v16bf a = {};
        #pragma unroll
        for (int p = 0; p < 8; ++p) {
            const int kp = 2 * p + 16 * half;
            a[2 * p]     = (__bf16)O[row * HID + k0 + kp];
            a[2 * p + 1] = (__bf16)O[row * HID + k0 + kp + 1];
        }
        v16bf b0 = {}, b1 = {};
        #pragma unroll
        for (int j = 0; j < 8; ++j) {
            const int kk = k0 + 2 * j + 16 * half;
            b0[2 * j]     = (__bf16)kW[kk * HID + n0 + r];
            b0[2 * j + 1] = (__bf16)kW[(kk + 1) * HID + n0 + r];
            b1[2 * j]     = (__bf16)kW[kk * HID + n0 + 16 + r];
            b1[2 * j + 1] = (__bf16)kW[(kk + 1) * HID + n0 + 16 + r];
        }
        acc0 = __builtin_amdgcn_wmma_f32_16x16x32_bf16(false, a, false, b0,
                                                       (short)0, acc0, false, false);
        acc1 = __builtin_amdgcn_wmma_f32_16x16x32_bf16(false, a, false, b1,
                                                       (short)0, acc1, false, false);
    }

    float part = 0.f;
    const int c0 = n0 + r;
    #pragma unroll
    for (int v = 0; v < 8; ++v) {
        part += tanhf(acc0[v] + kb[c0])      * q[c0];
        part += tanhf(acc1[v] + kb[c0 + 16]) * q[c0 + 16];
    }
    #pragma unroll
    for (int off = 16; off > 0; off >>= 1)
        part += __shfl_xor(part, off, 32);
    if (lane == 0)
        atomicAdd(score, part * invN);
}

// ---------------------------------------------------------------------------
// Per-node attention dot:  dot[n,h] = sum_d h[n, h*32+d] * att[h,d]
// ---------------------------------------------------------------------------
__global__ void node_dot(const float* __restrict__ H, const float* __restrict__ att,
                         float* __restrict__ dot, int N)
{
    const int idx = blockIdx.x * blockDim.x + threadIdx.x;
    if (idx >= N * HEADS) return;
    const int h = idx & 7;
    const float* hp = H + (size_t)(idx >> 3) * HID + h * DH;
    const float* ap = att + h * DH;
    float s = 0.f;
    #pragma unroll
    for (int d = 0; d < DH; ++d) s += hp[d] * ap[d];
    dot[idx] = s;
}

// ---------------------------------------------------------------------------
// Edge pass 1: score = leaky_relu(dot_src[src] + dot_dst[dst]); segment max
// ---------------------------------------------------------------------------
__global__ void edge_score_max(const float* __restrict__ ds, const float* __restrict__ dd,
                               const int* __restrict__ src, const int* __restrict__ dst,
                               float* __restrict__ aE, unsigned* __restrict__ segmax)
{
    const int idx = blockIdx.x * blockDim.x + threadIdx.x;
    if (idx >= NE * HEADS) return;
    const int e = idx >> 3, h = idx & 7;
    const int s = src[e], d = dst[e];
    float a = ds[s * HEADS + h] + dd[d * HEADS + h];
    a = (a > 0.f) ? a : 0.2f * a;            // leaky_relu, slope 0.2
    aE[idx] = a;
    atomicMax(&segmax[d * HEADS + h], fenc(a));
}

// ---------------------------------------------------------------------------
// Edge pass 2: e = exp(a - m[dst]); segment sum   (a overwritten with e)
// ---------------------------------------------------------------------------
__global__ void edge_exp_sum(const int* __restrict__ dst, const unsigned* __restrict__ segmax,
                             float* __restrict__ aE, float* __restrict__ segsum)
{
    const int idx = blockIdx.x * blockDim.x + threadIdx.x;
    if (idx >= NE * HEADS) return;
    const int e = idx >> 3, h = idx & 7;
    const int d = dst[e];
    const float m  = fdec(segmax[d * HEADS + h]);
    const float ev = expf(aE[idx] - m);
    aE[idx] = ev;
    atomicAdd(&segsum[d * HEADS + h], ev);
}

// ---------------------------------------------------------------------------
// Edge pass 3: wave per edge; out[dst,c] += h_src[src,c] * alpha[e, c>>5]
// ---------------------------------------------------------------------------
__global__ __launch_bounds__(256)
void edge_scatter(const float* __restrict__ H, const int* __restrict__ src,
                  const int* __restrict__ dst, const float* __restrict__ aE,
                  const float* __restrict__ segsum, float* __restrict__ Oout)
{
    const int e = blockIdx.x * 8 + (threadIdx.x >> 5);
    if (e >= NE) return;
    const int lane = threadIdx.x & 31;
    const int s = src[e], d = dst[e];
    float w[HEADS];
    #pragma unroll
    for (int k = 0; k < HEADS; ++k)
        w[k] = aE[e * HEADS + k] / (segsum[d * HEADS + k] + 1e-16f);
    #pragma unroll
    for (int k = 0; k < HEADS; ++k) {
        const int c = lane + 32 * k;
        atomicAdd(&Oout[(size_t)d * HID + c], H[(size_t)s * HID + c] * w[k]);
    }
}

// ---------------------------------------------------------------------------
// In-place ReLU + column sums (for global_add_pool)
// ---------------------------------------------------------------------------
__global__ __launch_bounds__(256)
void relu_colsum(float* __restrict__ O, float* __restrict__ colsum, int N)
{
    const int col = threadIdx.x;
    const int r0  = blockIdx.x * 16;
    float s = 0.f;
    for (int i = 0; i < 16; ++i) {
        const int r = r0 + i;
        if (r >= N) break;
        float v = O[(size_t)r * HID + col];
        v = v > 0.f ? v : 0.f;
        O[(size_t)r * HID + col] = v;
        s += v;
    }
    atomicAdd(&colsum[col], s);
}

// ---------------------------------------------------------------------------
// Finalize: metapath softmax (paper only; author has 1 metapath -> weight 1),
// pooled = weighted colsums, out = pooled @ lin_W + lin_b   ([2,16])
// ---------------------------------------------------------------------------
__global__ void finalize(const float* __restrict__ colw, const float* __restrict__ colc,
                         const float* __restrict__ colr, const float* __restrict__ scores,
                         const float* __restrict__ linW, const float* __restrict__ linb,
                         float* __restrict__ out)
{
    const int tid = threadIdx.x;
    if (tid >= 2 * OUTF) return;
    const float s0 = scores[0], s1 = scores[1];
    const float mx = fmaxf(s0, s1);
    const float e0 = expf(s0 - mx), e1 = expf(s1 - mx);
    const float a0 = e0 / (e0 + e1), a1 = e1 / (e0 + e1);
    const int t = tid >> 4;        // 0 = author, 1 = paper
    const int o = tid & 15;
    float acc = linb[o];
    for (int c = 0; c < HID; ++c) {
        const float p = (t == 0) ? colr[c] : (a0 * colw[c] + a1 * colc[c]);
        acc += p * linW[c * OUTF + o];
    }
    out[tid] = acc;
}

// ---------------------------------------------------------------------------
extern "C" void kernel_launch(void* const* d_in, const int* in_sizes, int n_in,
                              void* d_out, int out_size, void* d_ws, size_t ws_size,
                              hipStream_t stream)
{
    const float* x_author = (const float*)d_in[0];
    const float* x_paper  = (const float*)d_in[1];
    const float* W_author = (const float*)d_in[2];
    const float* b_author = (const float*)d_in[3];
    const float* W_paper  = (const float*)d_in[4];
    const float* b_paper  = (const float*)d_in[5];
    const float* att_sw   = (const float*)d_in[6];
    const float* att_dw   = (const float*)d_in[7];
    const float* att_sc   = (const float*)d_in[8];
    const float* att_dc   = (const float*)d_in[9];
    const float* att_sr   = (const float*)d_in[10];
    const float* att_dr   = (const float*)d_in[11];
    const float* k_W      = (const float*)d_in[12];
    const float* k_b      = (const float*)d_in[13];
    const float* q        = (const float*)d_in[14];
    const float* lin_W    = (const float*)d_in[15];
    const float* lin_b    = (const float*)d_in[16];
    const int* ew_src = (const int*)d_in[17];
    const int* ew_dst = (const int*)d_in[18];
    const int* ec_src = (const int*)d_in[19];
    const int* ec_dst = (const int*)d_in[20];
    const int* er_src = (const int*)d_in[21];
    const int* er_dst = (const int*)d_in[22];
    float* out = (float*)d_out;

    // ---- workspace layout (floats) ----
    float* ws = (float*)d_ws;
    float* h_a     = ws;                             // NA*HID
    float* h_p     = h_a     + (size_t)NA * HID;     // NPAP*HID
    float* dot_aw  = h_p     + (size_t)NPAP * HID;   // NA*8
    float* dot_pw  = dot_aw  + (size_t)NA * HEADS;   // NPAP*8
    float* dot_pcs = dot_pw  + (size_t)NPAP * HEADS; // NPAP*8
    float* dot_pcd = dot_pcs + (size_t)NPAP * HEADS; // NPAP*8
    float* dot_prs = dot_pcd + (size_t)NPAP * HEADS; // NPAP*8
    float* dot_ard = dot_prs + (size_t)NPAP * HEADS; // NA*8
    float* aE      = dot_ard + (size_t)NA * HEADS;   // NE*8 (reused per type)
    float* segmaxf = aE      + (size_t)NE * HEADS;   // NPAP*8 (uint encoded)
    float* segsum  = segmaxf + (size_t)NPAP * HEADS; // NPAP*8
    float* o_w     = segsum  + (size_t)NPAP * HEADS; // NPAP*HID
    float* o_c     = o_w     + (size_t)NPAP * HID;   // NPAP*HID
    float* o_r     = o_c     + (size_t)NPAP * HID;   // NA*HID
    float* colsum_w = o_r      + (size_t)NA * HID;   // HID
    float* colsum_c = colsum_w + HID;                // HID
    float* colsum_r = colsum_c + HID;                // HID
    float* scores   = colsum_r + HID;                // 2
    unsigned* segmax = (unsigned*)segmaxf;

    // zero all accumulation buffers (o_w..scores are contiguous)
    const size_t accFloats = (size_t)NPAP * HID * 2 + (size_t)NA * HID + 3 * HID + 2;
    hipMemsetAsync(o_w, 0, accFloats * sizeof(float), stream);

    // ---- node-type projections (WMMA, async-LDS staged weights) ----
    gemm_bias_wmma<<<NA / 16, 256, 0, stream>>>(x_author, W_author, b_author, h_a, NA);
    gemm_bias_wmma<<<NPAP / 16, 256, 0, stream>>>(x_paper, W_paper, b_paper, h_p, NPAP);

    // ---- per-node attention dots ----
    const int dbA = (NA * HEADS + 255) / 256;
    const int dbP = (NPAP * HEADS + 255) / 256;
    node_dot<<<dbA, 256, 0, stream>>>(h_a, att_sw, dot_aw, NA);
    node_dot<<<dbP, 256, 0, stream>>>(h_p, att_dw, dot_pw, NPAP);
    node_dot<<<dbP, 256, 0, stream>>>(h_p, att_sc, dot_pcs, NPAP);
    node_dot<<<dbP, 256, 0, stream>>>(h_p, att_dc, dot_pcd, NPAP);
    node_dot<<<dbP, 256, 0, stream>>>(h_p, att_sr, dot_prs, NPAP);
    node_dot<<<dbA, 256, 0, stream>>>(h_a, att_dr, dot_ard, NA);

    const int ebl = (NE * HEADS) / 256;   // 8000
    const int sbl = NE / 8;               // 32000
    const size_t segBytes = (size_t)NPAP * HEADS * 2 * sizeof(float);

    // ---- edge type: writes (author -> paper) ----
    hipMemsetAsync(segmaxf, 0, segBytes, stream);
    edge_score_max<<<ebl, 256, 0, stream>>>(dot_aw, dot_pw, ew_src, ew_dst, aE, segmax);
    edge_exp_sum<<<ebl, 256, 0, stream>>>(ew_dst, segmax, aE, segsum);
    edge_scatter<<<sbl, 256, 0, stream>>>(h_a, ew_src, ew_dst, aE, segsum, o_w);

    // ---- edge type: cites (paper -> paper) ----
    hipMemsetAsync(segmaxf, 0, segBytes, stream);
    edge_score_max<<<ebl, 256, 0, stream>>>(dot_pcs, dot_pcd, ec_src, ec_dst, aE, segmax);
    edge_exp_sum<<<ebl, 256, 0, stream>>>(ec_dst, segmax, aE, segsum);
    edge_scatter<<<sbl, 256, 0, stream>>>(h_p, ec_src, ec_dst, aE, segsum, o_c);

    // ---- edge type: rev_writes (paper -> author) ----
    hipMemsetAsync(segmaxf, 0, segBytes, stream);
    edge_score_max<<<ebl, 256, 0, stream>>>(dot_prs, dot_ard, er_src, er_dst, aE, segmax);
    edge_exp_sum<<<ebl, 256, 0, stream>>>(er_dst, segmax, aE, segsum);
    edge_scatter<<<sbl, 256, 0, stream>>>(h_p, er_src, er_dst, aE, segsum, o_r);

    // ---- ReLU + pooled column sums ----
    relu_colsum<<<(NPAP + 15) / 16, 256, 0, stream>>>(o_w, colsum_w, NPAP);
    relu_colsum<<<(NPAP + 15) / 16, 256, 0, stream>>>(o_c, colsum_c, NPAP);
    relu_colsum<<<(NA + 15) / 16, 256, 0, stream>>>(o_r, colsum_r, NA);

    // ---- semantic attention scores (WMMA fused tanh·q); author metapath
    //      softmax over a single element is identically 1 -> skipped ----
    semantic_score_wmma<<<NPAP / 16, 256, 0, stream>>>(o_w, k_W, k_b, q, &scores[0],
                                                       1.0f / NPAP, NPAP);
    semantic_score_wmma<<<NPAP / 16, 256, 0, stream>>>(o_c, k_W, k_b, q, &scores[1],
                                                       1.0f / NPAP, NPAP);

    // ---- final pooled linear head ----
    finalize<<<1, 32, 0, stream>>>(colsum_w, colsum_c, colsum_r, scores,
                                   lin_W, lin_b, out);
}
